// LiquidNeuronLayer_30743375905304
// MI455X (gfx1250) — compile-verified, matrix operand order checked by reading, not run on hardware
//
#include <hip/hip_runtime.h>

// ---- problem dims (fixed by reference) ----
constexpr int B_ = 16, T_ = 2048, D_ = 256, H_ = 512;
constexpr int M_ = B_ * T_;           // 32768 rows for phase-1 GEMMs
#define DT_CONST 0.1f

typedef _Float16 v16h __attribute__((ext_vector_type(16)));
typedef _Float16 v8h  __attribute__((ext_vector_type(8)));
typedef float    v8f  __attribute__((ext_vector_type(8)));

// ---------------------------------------------------------------------------
// Fragment loader (16-bit A/B layout for V_WMMA_*_16X16X32):
//   lane L (L<16): row = L, holds K = kb..kb+7 (VGPR0-3) and kb+16..kb+23 (VGPR4-7)
//   lane L+16:     row = L, with kb += 8
// Per lane: two contiguous 8-element (16B) chunks at k0 and k0+16.
// B fragments use the same pattern with "row" = output column (W row-major,
// out = X @ W^T  =>  B[k][n] = W[n][k]).
// ---------------------------------------------------------------------------
template <typename PT>
__device__ inline v16h load_frag16(const PT* base, int row, int ld, int kbase, int lane) {
  const PT* p = base + row * ld + kbase + ((lane >> 4) << 3);
  v8h lo = *(const v8h*)(p);
  v8h hi = *(const v8h*)(p + 16);
  return __builtin_shufflevector(lo, hi, 0, 1, 2, 3, 4, 5, 6, 7,
                                 8, 9, 10, 11, 12, 13, 14, 15);
}

__device__ inline v8f wmma_f16(v16h a, v16h b, v8f c) {
  return __builtin_amdgcn_wmma_f32_16x16x32_f16(false, a, false, b, (short)0, c,
                                                false, false);
}

__device__ inline float fast_tanh(float x) {
  // tanh(x) = 1 - 2/(exp(2x)+1); saturates correctly for large |x|
  float e = __expf(2.0f * x);
  return 1.0f - 2.0f / (e + 1.0f);
}

// ---------------------------------------------------------------------------
// f32 -> f16 conversion (grid-stride)
// ---------------------------------------------------------------------------
__global__ void cvt_f32_f16(const float* __restrict__ src, _Float16* __restrict__ dst, int n) {
  int i = blockIdx.x * blockDim.x + threadIdx.x;
  int stride = gridDim.x * blockDim.x;
  for (; i < n; i += stride) dst[i] = (_Float16)src[i];
}

// ---------------------------------------------------------------------------
// Phase 1: xw = x @ Wx^T + b  -> d_out  (overwritten in place by the scan)
//          a  = DT / (softplus(x @ tau_w^T + tau_b) + 0.01) -> ws
// One wave per 16x16 output tile; A (x) fragments shared by both GEMMs.
// ---------------------------------------------------------------------------
__global__ __launch_bounds__(256)
void phase1_gemm(const _Float16* __restrict__ xh,
                 const _Float16* __restrict__ wxh,
                 const _Float16* __restrict__ twh,
                 const float* __restrict__ bvec,
                 const float* __restrict__ tbvec,
                 float* __restrict__ xw_out,
                 float* __restrict__ a_out) {
  const int lane = threadIdx.x & 31;
  const int wave = threadIdx.x >> 5;
  const int tile = blockIdx.x * (blockDim.x >> 5) + wave;
  const int NT = H_ / 16;               // 32 column tiles
  const int mt = tile / NT;
  const int nt = tile - mt * NT;
  const int row0 = mt * 16;
  const int n0 = nt * 16;

  const int arow = row0 + (lane & 15);
  const int brow = n0 + (lane & 15);

  v8f accX = {};
  v8f accT = {};
#pragma unroll
  for (int kt = 0; kt < D_; kt += 32) {
    v16h af = load_frag16(xh, arow, D_, kt, lane);
    v16h bx = load_frag16(wxh, brow, D_, kt, lane);
    v16h bt = load_frag16(twh, brow, D_, kt, lane);
    accX = wmma_f16(af, bx, accX);
    accT = wmma_f16(af, bt, accT);
  }

  // C/D layout: lane<16 -> n=lane, m=vgpr; lane>=16 -> n=lane-16, m=vgpr+8
  const int n = n0 + (lane & 15);
  const int mhi = (lane >> 4) * 8;
  const float bn = bvec[n];
  const float tbn = tbvec[n];
#pragma unroll
  for (int v = 0; v < 8; ++v) {
    int m = row0 + mhi + v;
    float xw = accX[v] + bn;
    float tr = accT[v] + tbn;
    float sp = fmaxf(tr, 0.0f) + __logf(1.0f + __expf(-fabsf(tr)));
    xw_out[m * H_ + n] = xw;
    a_out[m * H_ + n] = DT_CONST / (sp + 0.01f);
  }
}

// ---------------------------------------------------------------------------
// Phase 2: sequential scan. ONE workgroup, 32 waves; wave w owns H-columns
// [16w, 16w+16). Low register pressure by construction (~100 VGPRs):
//   - Wh[:, 0:256]  as f16 in LDS (row stride 264 -> conflict-free fragments)
//   - Wh[:, 256:512] streamed from global each step (L2-resident, 256 KB)
//   - h (16x512 f16) in LDS, padded row stride 520
// Per step: 16 wmma on 2 accumulators, tanh, liquid update, h -> LDS + d_out.
// ---------------------------------------------------------------------------
constexpr int HSTRIDE = 520;  // hbuf row stride in f16 (512 + 8 pad)
constexpr int WSTRIDE = 264;  // whl row stride in f16 (256 + 8 pad)
constexpr int KHALF = 256;

__global__ __launch_bounds__(1024)
void scan_kernel(const _Float16* __restrict__ whh,
                 const float* __restrict__ a_in,
                 float* __restrict__ out) {
  __shared__ _Float16 whl[H_ * WSTRIDE];     // 264 KB (CDNA5: 320 KB LDS/WGP)
  __shared__ _Float16 hbuf[16 * HSTRIDE];    // 16.25 KB

  const int tid = threadIdx.x;
  const int lane = tid & 31;
  const int wave = tid >> 5;   // 0..31

  // Cooperative copy: Wh[n][0:256] -> whl (padded), 16B chunks.
  for (int idx = tid; idx < H_ * (KHALF / 8); idx += 1024) {
    int nr = idx >> 5;               // row (output column)
    int c = (idx & 31) << 3;         // f16 offset within K-half
    *(v8h*)(&whl[nr * WSTRIDE + c]) = *(const v8h*)(&whh[nr * H_ + c]);
  }
  __syncthreads();

  const int n = wave * 16 + (lane & 15);   // this lane's output column
  const int mhi = (lane >> 4) * 8;

  v8f h;
  // ---- t = 0: h = a * tanh(xw) (h0 == 0) ----
#pragma unroll
  for (int v = 0; v < 8; ++v) {
    int m = mhi + v;                       // batch row
    int idx = (m * T_) * H_ + n;           // t = 0
    float z = fast_tanh(out[idx]);
    h[v] = a_in[idx] * z;
    out[idx] = h[v];
    hbuf[m * HSTRIDE + n] = (_Float16)h[v];
  }
  __syncthreads();

  for (int t = 1; t < T_; ++t) {
    // Launder the Wh pointer so the upper-K B-fragments (loop-invariant)
    // are re-loaded per iteration instead of hoisted into 64+ registers.
    const _Float16* whg = whh;
    asm volatile("" : "+v"(whg));

    float xw[8], av[8];
#pragma unroll
    for (int v = 0; v < 8; ++v) {
      int m = mhi + v;
      int idx = (m * T_ + t) * H_ + n;
      xw[v] = out[idx];
      av[v] = a_in[idx];
    }

    v8f acc0 = {}, acc1 = {};
#pragma unroll
    for (int j = 0; j < 8; ++j) {
      v16h a0 = load_frag16(hbuf, (lane & 15), HSTRIDE, j * 32, lane);
      v16h b0 = load_frag16(whl, n, WSTRIDE, j * 32, lane);
      acc0 = wmma_f16(a0, b0, acc0);
      v16h a1 = load_frag16(hbuf, (lane & 15), HSTRIDE, KHALF + j * 32, lane);
      v16h b1 = load_frag16(whg, n, H_, KHALF + j * 32, lane);
      acc1 = wmma_f16(a1, b1, acc1);
    }

#pragma unroll
    for (int v = 0; v < 8; ++v) {
      float z = fast_tanh(acc0[v] + acc1[v] + xw[v]);
      h[v] = h[v] + av[v] * (z - h[v]);    // h += (DT/tau)*(z-h)
    }

    __syncthreads();  // all waves done reading hbuf(t-1)
#pragma unroll
    for (int v = 0; v < 8; ++v) {
      int m = mhi + v;
      hbuf[m * HSTRIDE + n] = (_Float16)h[v];
      out[(m * T_ + t) * H_ + n] = h[v];
    }
    __syncthreads();  // h(t) visible before step t+1 reads
  }
}

// ---------------------------------------------------------------------------
// Launch
// ---------------------------------------------------------------------------
extern "C" void kernel_launch(void* const* d_in, const int* in_sizes, int n_in,
                              void* d_out, int out_size, void* d_ws, size_t ws_size,
                              hipStream_t stream) {
  const float* x   = (const float*)d_in[0];  // (B,T,D)
  const float* Wx  = (const float*)d_in[1];  // (H,D)
  const float* Wh  = (const float*)d_in[2];  // (H,H)
  const float* b   = (const float*)d_in[3];  // (H)
  const float* tw  = (const float*)d_in[4];  // (H,D)
  const float* tb  = (const float*)d_in[5];  // (H)
  float* out = (float*)d_out;                // (B,T,H)

  char* ws = (char*)d_ws;
  float*    tauA = (float*)ws;                                   // M*H f32 (64 MB)
  _Float16* xh   = (_Float16*)(ws + (size_t)M_ * H_ * 4);        // M*D f16 (16 MB)
  _Float16* wxh  = xh + (size_t)M_ * D_;                         // H*D f16
  _Float16* twh  = wxh + (size_t)H_ * D_;                        // H*D f16
  _Float16* whh  = twh + (size_t)H_ * D_;                        // H*H f16

  // f32 -> f16 staging
  cvt_f32_f16<<<2048, 256, 0, stream>>>(x, xh, M_ * D_);
  cvt_f32_f16<<<128, 256, 0, stream>>>(Wx, wxh, H_ * D_);
  cvt_f32_f16<<<128, 256, 0, stream>>>(tw, twh, H_ * D_);
  cvt_f32_f16<<<256, 256, 0, stream>>>(Wh, whh, H_ * H_);

  // Phase 1: 2048 * 32 tiles, 8 waves (tiles) per 256-thread block
  phase1_gemm<<<(M_ / 16) * (H_ / 16) / 8, 256, 0, stream>>>(xh, wxh, twh, b, tb,
                                                             out, tauA);

  // Phase 2: single persistent workgroup (latency-bound recurrence)
  scan_kernel<<<1, 1024, 0, stream>>>(whh, tauA, out);
}
